// Pairwise_45707041964432
// MI455X (gfx1250) — matrix operand order repaired and checked by reading, not compiled, not run on hardware
//
#include <hip/hip_runtime.h>

typedef _Float16 h16;
typedef __attribute__((ext_vector_type(16))) _Float16 v16h;
typedef __attribute__((ext_vector_type(8)))  float    v8f;
typedef __attribute__((ext_vector_type(4)))  int      i4v;

union V16H { v16h v; h16 h[16]; uint4 q[2]; };

#define B_   32
#define N_   36
#define D_   2048
#define R_   512
#define K_   1024                // fused rank: [feat 512 | coord 512]
#define ROWS (B_*N_)             // 1152
#define MT   (ROWS/16)           // 72 row tiles (exact, no padding)
#define DT   (D_/128)            // 16 d-tiles of 128
#define KC_M (K_/32)             // 32 k-chunks in main GEMM
#define KC_P (D_/32)             // 64 k-chunks in projection GEMM

// LDS layout for main kernel (bytes)
#define LDS_B_OFF 0              // [kc][wave][lane] * 32B  = 262144
#define LDS_A_OFF 262144         // [kc][lane] * 32B        = 32768
#define LDS_V_OFF 294912         // 2 rows * 1024 f32       = 8192
#define LDS_TOTAL 303104

// ---- optional CDNA5 async global->LDS path (guarded; sync fallback kept) ----
#if defined(__has_builtin)
#  if __has_builtin(__builtin_amdgcn_global_load_async_to_lds_b128) && \
      __has_builtin(__builtin_amdgcn_s_wait_asynccnt)
#    define USE_ASYNC_LDS 1
#  endif
#endif
#ifndef USE_ASYNC_LDS
#  define USE_ASYNC_LDS 0
#endif

#if USE_ASYNC_LDS
#define GLD_ASYNC_B128(gsrc, ldst, off)                                   \
  __builtin_amdgcn_global_load_async_to_lds_b128(                         \
      (__attribute__((address_space(1))) i4v*)(gsrc),                     \
      (__attribute__((address_space(3))) i4v*)(ldst), (off), 0)
#define WAIT_ASYNC() __builtin_amdgcn_s_wait_asynccnt(0)
#endif

// ---------------- prep kernels ----------------
__global__ void prep_mmh(const float* __restrict__ mm, h16* __restrict__ mmh, int n) {
  int i = blockIdx.x * blockDim.x + threadIdx.x;
  if (i < n) mmh[i] = (h16)mm[i];
}

// Wth[n][k] = (n<512 ? U_feat[k][n] : V_feat[k][n-512]),  n<1024, k<2048
__global__ void prep_wth(const float* __restrict__ Uf, const float* __restrict__ Vf,
                         h16* __restrict__ Wth) {
  int i = blockIdx.x * blockDim.x + threadIdx.x;
  if (i >= 1024 * 2048) return;
  int nn = i >> 11, k = i & 2047;
  float v = (nn < 512) ? Uf[k * 512 + nn] : Vf[k * 512 + (nn - 512)];
  Wth[i] = (h16)v;
}

// Pth[d][k] = (k<512 ? P_feat[k][d] : P_coord[k-512][d]),  d<2048, k<1024
__global__ void prep_pth(const float* __restrict__ Pf, const float* __restrict__ Pc,
                         h16* __restrict__ Pth) {
  int i = blockIdx.x * blockDim.x + threadIdx.x;
  if (i >= 2048 * 1024) return;
  int d = i >> 10, k = i & 1023;
  float v = (k < 512) ? Pf[k * 2048 + d] : Pc[(k - 512) * 2048 + d];
  Pth[i] = (h16)v;
}

// coord projections (K=4 dots, f32): Ubuf[row][512+r]=uc, Vbuf[row][512+r]=vc
__global__ void prep_uvc(const float* __restrict__ coords, const float* __restrict__ Uc,
                         const float* __restrict__ Vc, float* __restrict__ Ubuf,
                         float* __restrict__ Vbuf) {
  int i = blockIdx.x * blockDim.x + threadIdx.x;
  if (i >= ROWS * 512) return;
  int row = i >> 9, r = i & 511;
  float c0 = coords[row*4+0], c1 = coords[row*4+1], c2 = coords[row*4+2], c3 = coords[row*4+3];
  float uc = c0*Uc[r] + c1*Uc[512+r] + c2*Uc[1024+r] + c3*Uc[1536+r];
  float vc = c0*Vc[r] + c1*Vc[512+r] + c2*Vc[1024+r] + c3*Vc[1536+r];
  Ubuf[row*1024 + 512 + r] = uc;
  Vbuf[row*1024 + 512 + r] = vc;
}

// ---------------- projection GEMM: [1152,2048] x Wth^T -> uf|vf ----------------
// grid (72, 8), block 256. Wave w handles n-tile = ngrp*128 + w*16.
__global__ __launch_bounds__(256)
void proj_gemm(const h16* __restrict__ mmh, const h16* __restrict__ Wth,
               float* __restrict__ Ubuf, float* __restrict__ Vbuf) {
  int mtile = blockIdx.x, ngrp = blockIdx.y;
  int lane = threadIdx.x & 31, w = threadIdx.x >> 5;
  int l15 = lane & 15, hi = lane >> 4;
  int row  = mtile * 16 + l15;
  int ncol = ngrp * 128 + w * 16 + l15;
  const h16* arow = mmh + (size_t)row  * 2048;
  const h16* brow = Wth + (size_t)ncol * 2048;

  v8f c = {0,0,0,0,0,0,0,0};
  for (int kc = 0; kc < KC_P; ++kc) {
    int k0 = kc * 32;
    V16H a, b;
    // A fragment: lane<16 holds K {0..7,16..23}, lane>=16 holds {8..15,24..31}
    a.q[0] = *(const uint4*)(arow + k0 + (hi ? 8 : 0));
    a.q[1] = *(const uint4*)(arow + k0 + (hi ? 8 : 0) + 16);
    // B fragment: lane<16 holds K {0..7,16..23}, lane>=16 holds {8..15,24..31}
    const uint4* b0 = (const uint4*)(brow + k0 + (hi ? 8 : 0));
    b.q[0] = b0[0]; b.q[1] = b0[2];
    c = __builtin_amdgcn_wmma_f32_16x16x32_f16(false, a.v, false, b.v,
                                               (short)0, c, false, false);
  }
#pragma unroll
  for (int r = 0; r < 8; ++r) {
    int m = mtile * 16 + (hi ? r + 8 : r);
    float v = c[r];
    if (ncol < 512) Ubuf[(size_t)m * 1024 + ncol]         = v;   // uf
    else            Vbuf[(size_t)m * 1024 + (ncol - 512)] = v;   // vf
  }
}

// ---------------- main fused pairwise kernel ----------------
// grid (72, 16), block 256, dynamic LDS = LDS_TOTAL
__global__ __launch_bounds__(256)
void pairwise_main(const float* __restrict__ Ubuf, const float* __restrict__ Vbuf,
                   const h16* __restrict__ Pth, const float* __restrict__ mm,
                   float* __restrict__ out) {
  extern __shared__ char smem[];
  char*  Bs = smem + LDS_B_OFF;
  char*  As = smem + LDS_A_OFF;
  float* Vs = (float*)(smem + LDS_V_OFF);

  int tile = blockIdx.x;           // row tile 0..71
  int dt   = blockIdx.y;           // d tile 0..15
  int tid  = threadIdx.x;
  int lane = tid & 31, w = tid >> 5;
  int l15 = lane & 15, hi = lane >> 4;
  int dbase = dt * 128;

  // ---- stage B tile (P^T) into LDS in fragment order: contiguous 32B per (kc,w,lane)
  {
    int d = dbase + w * 16 + l15;
    const h16* prow = Pth + (size_t)d * 1024;
    for (int kc = 0; kc < KC_M; ++kc) {
      const h16* src = prow + kc * 32 + (hi ? 8 : 0);
      char* dst = Bs + (((kc * 8 + w) * 32 + lane) << 5);
#if USE_ASYNC_LDS
      GLD_ASYNC_B128(src, dst, 0);
      GLD_ASYNC_B128(src + 16, dst + 16, 0);
#else
      ((uint4*)dst)[0] = ((const uint4*)src)[0];
      ((uint4*)dst)[1] = ((const uint4*)src)[2];
#endif
    }
  }

  // ---- persistent u registers: wave w owns A-build for kc = w*4 .. w*4+3
  int grow = tile * 16 + l15;               // A row for this lane
  const float* urow = Ubuf + (size_t)grow * 1024;
  int kc0 = w * 4;
  float ur[4][16];
#pragma unroll
  for (int q = 0; q < 4; ++q) {
    int base = (kc0 + q) * 32 + (hi ? 8 : 0);
#pragma unroll
    for (int e = 0; e < 8; ++e) { ur[q][e] = urow[base + e]; ur[q][8 + e] = urow[base + 16 + e]; }
  }

  int b_lo = (tile * 16) / N_;
  int b_hi = (tile * 16 + 15) / N_;         // tiles span at most 2 batches (36 > 16)
  int vsel = ((grow / N_) != b_lo) ? 1024 : 0;

  v8f cmax = {-3.0e38f,-3.0e38f,-3.0e38f,-3.0e38f,-3.0e38f,-3.0e38f,-3.0e38f,-3.0e38f};

  for (int j = 0; j < N_; ++j) {
    __syncthreads();  // previous iteration finished reading As/Vs
    // stage v_j rows for both batches touched by this tile (2 x 1024 f32)
    {
      const float* v0 = Vbuf + (size_t)(b_lo * N_ + j) * 1024;
      const float* v1 = Vbuf + (size_t)(b_hi * N_ + j) * 1024;
#if USE_ASYNC_LDS
      GLD_ASYNC_B128(v0 + tid * 4, (char*)Vs + tid * 16, 0);
      GLD_ASYNC_B128(v1 + tid * 4, (char*)Vs + 4096 + tid * 16, 0);
      WAIT_ASYNC();   // own async copies (incl. B tile at j==0) complete
#else
      ((float4*)Vs)[tid]       = ((const float4*)v0)[tid];
      ((float4*)Vs)[256 + tid] = ((const float4*)v1)[tid];
#endif
    }
    __syncthreads(); // all waves' staging complete
    // build A = f16(relu(u .* v_j)) in fragment order
#pragma unroll
    for (int q = 0; q < 4; ++q) {
      int kc = kc0 + q;
      int base = vsel + kc * 32 + (hi ? 8 : 0);
      V16H av;
#pragma unroll
      for (int e = 0; e < 8; ++e) {
        float p0 = ur[q][e]     * Vs[base + e];
        float p1 = ur[q][8 + e] * Vs[base + 16 + e];
        av.h[e]     = (h16)(p0 > 0.f ? p0 : 0.f);
        av.h[8 + e] = (h16)(p1 > 0.f ? p1 : 0.f);
      }
      uint4* dst = (uint4*)(As + ((kc * 32 + lane) << 5));
      dst[0] = av.q[0]; dst[1] = av.q[1];
    }
    __syncthreads();
    // GEMM: 16 x 1024 (A) x 1024 x 16 (this wave's n-tile), f32 accumulate
    v8f c = {0,0,0,0,0,0,0,0};
    for (int kc = 0; kc < KC_M; ++kc) {
      V16H a, b;
      const uint4* ap = (const uint4*)(As + ((kc * 32 + lane) << 5));
      a.q[0] = ap[0]; a.q[1] = ap[1];
      const uint4* bp = (const uint4*)(Bs + (((kc * 8 + w) * 32 + lane) << 5));
      b.q[0] = bp[0]; b.q[1] = bp[1];
      c = __builtin_amdgcn_wmma_f32_16x16x32_f16(false, a.v, false, b.v,
                                                 (short)0, c, false, false);
    }
#pragma unroll
    for (int r = 0; r < 8; ++r) cmax[r] = fmaxf(cmax[r], c[r]);
  }

  // store with residual
  int dcol = dbase + w * 16 + l15;
#pragma unroll
  for (int r = 0; r < 8; ++r) {
    int m = tile * 16 + (hi ? r + 8 : r);
    size_t idx = (size_t)m * 2048 + dcol;
    out[idx] = cmax[r] + mm[idx];
  }
}

// ---------------- launcher ----------------
extern "C" void kernel_launch(void* const* d_in, const int* in_sizes, int n_in,
                              void* d_out, int out_size, void* d_ws, size_t ws_size,
                              hipStream_t stream) {
  const float* mm     = (const float*)d_in[0];
  const float* coords = (const float*)d_in[1];
  const float* U_feat = (const float*)d_in[2];
  const float* V_feat = (const float*)d_in[3];
  const float* P_feat = (const float*)d_in[4];
  const float* U_coord= (const float*)d_in[5];
  const float* V_coord= (const float*)d_in[6];
  const float* P_coord= (const float*)d_in[7];
  float* out = (float*)d_out;

  char* ws = (char*)d_ws;
  h16*   mmh  = (h16*)(ws + 0);                      // 1152*2048 f16 = 4718592 B
  h16*   Wth  = (h16*)(ws + 4718592);                // 1024*2048 f16 = 4194304 B
  h16*   Pth  = (h16*)(ws + 8912896);                // 2048*1024 f16 = 4194304 B
  float* Ubuf = (float*)(ws + 13107200);             // 1152*1024 f32 = 4718592 B
  float* Vbuf = (float*)(ws + 17825792);             // 1152*1024 f32 = 4718592 B
  (void)in_sizes; (void)n_in; (void)out_size; (void)ws_size;

  const int NMM = ROWS * D_;  // 2359296
  prep_mmh<<<(NMM + 255) / 256, 256, 0, stream>>>(mm, mmh, NMM);
  prep_wth<<<(1024 * 2048) / 256, 256, 0, stream>>>(U_feat, V_feat, Wth);
  prep_pth<<<(2048 * 1024) / 256, 256, 0, stream>>>(P_feat, P_coord, Pth);
  prep_uvc<<<(ROWS * 512) / 256, 256, 0, stream>>>(coords, U_coord, V_coord, Ubuf, Vbuf);

  proj_gemm<<<dim3(MT, 8), 256, 0, stream>>>(mmh, Wth, Ubuf, Vbuf);

  pairwise_main<<<dim3(MT, DT), 256, LDS_TOTAL, stream>>>(Ubuf, Vbuf, Pth, mm, out);
}
